// GRELEN_41240275976730
// MI455X (gfx1250) — matrix operand order repaired and checked by reading, not compiled, not run on hardware
//
#include <hip/hip_runtime.h>
#include <hip/hip_bf16.h>
#include <math.h>

typedef __attribute__((ext_vector_type(16))) _Float16 v16h;
typedef __attribute__((ext_vector_type(8)))  _Float16 v8h;
typedef __attribute__((ext_vector_type(4)))  _Float16 v4h;
typedef __attribute__((ext_vector_type(8)))  float    v8f;

#define GB   16
#define GN   1024
#define GU   64
#define TWOU 128
#define GTAU 0.5f
#define GEPS 1e-10f

// ---------------------------------------------------------------------------
// CDNA5 async global->LDS copy (no VGPR round-trip, tracked by ASYNCcnt).
// lds  = workgroup-relative LDS byte offset (low 32 bits of generic pointer)
// ---------------------------------------------------------------------------
__device__ __forceinline__ void async_ld_b128(unsigned lds, const void* g)
{
    unsigned long long ga = (unsigned long long)g;
    asm volatile("global_load_async_to_lds_b128 %0, %1, off"
                 :: "v"(lds), "v"(ga)
                 : "memory");
}
__device__ __forceinline__ void wait_asynccnt0()
{
    asm volatile("s_wait_asynccnt 0x0" ::: "memory");
}
__device__ __forceinline__ unsigned lds_off(const void* p)
{
    return (unsigned)(unsigned long long)p;   // addrspace(3) offset in low bits
}

// ---------------------------------------------------------------------------
// Kernel 1: fold diffusion weight columns (K taps: [x, Ax, Ax]) into f16
// B-matrices stored (K x N) row-major, plus bias sums.
// ---------------------------------------------------------------------------
__global__ __launch_bounds__(256)
void k_prep(const float* __restrict__ W0, const float* __restrict__ b0,
            const float* __restrict__ W1, const float* __restrict__ b1,
            const float* __restrict__ Wc0, const float* __restrict__ bc0,
            const float* __restrict__ Wc1, const float* __restrict__ bc1,
            _Float16* __restrict__ Wxa, _Float16* __restrict__ W0b,
            _Float16* __restrict__ W1b, _Float16* __restrict__ Wcxa,
            _Float16* __restrict__ Wc0b, _Float16* __restrict__ Wc1b,
            float* __restrict__ bsum, float* __restrict__ bcsum)
{
    const int tid = threadIdx.x;
    for (int idx = tid; idx < 128 * 128; idx += 256) {
        int k = idx >> 7, o = idx & 127;              // B[k][o] = W[o][3k..]^T
        Wxa[idx] = (_Float16)(W0[o * 384 + 3 * k] + W1[o * 384 + 3 * k]);
        W0b[idx] = (_Float16)(W0[o * 384 + 3 * k + 1] + W0[o * 384 + 3 * k + 2]);
        W1b[idx] = (_Float16)(W1[o * 384 + 3 * k + 1] + W1[o * 384 + 3 * k + 2]);
    }
    for (int idx = tid; idx < 128 * 64; idx += 256) {
        int k = idx >> 6, o = idx & 63;
        Wcxa[idx] = (_Float16)(Wc0[o * 384 + 3 * k] + Wc1[o * 384 + 3 * k]);
        Wc0b[idx] = (_Float16)(Wc0[o * 384 + 3 * k + 1] + Wc0[o * 384 + 3 * k + 2]);
        Wc1b[idx] = (_Float16)(Wc1[o * 384 + 3 * k + 1] + Wc1[o * 384 + 3 * k + 2]);
    }
    if (tid < 128) bsum[tid] = b0[tid] + b1[tid];
    if (tid < 64)  bcsum[tid] = bc0[tid] + bc1[tid];
}

// ---------------------------------------------------------------------------
// Kernel 2: pack X1 = [x, h] (f16) and the static half of X2 = [x, .]
// ---------------------------------------------------------------------------
__global__ __launch_bounds__(256)
void k_pack(const float* __restrict__ inputs, const float* __restrict__ hx,
            _Float16* __restrict__ X1, _Float16* __restrict__ X2)
{
    const int gid = blockIdx.x * 256 + threadIdx.x;  // (b*N+i)*64 + u
    const int row = gid >> 6, uu = gid & 63;
    _Float16 xv = (_Float16)inputs[gid];
    X1[(size_t)row * TWOU + uu]      = xv;
    X1[(size_t)row * TWOU + 64 + uu] = (_Float16)hx[gid];
    X2[(size_t)row * TWOU + uu]      = xv;
}

// ---------------------------------------------------------------------------
// Kernel 3: adjacency (gumbel-softmax comp 0) + row sums -> invd0
// One block per (b, row). 256 threads * 4 cols each.
// ---------------------------------------------------------------------------
__global__ __launch_bounds__(256)
void k_adj(const float* __restrict__ logits, const float* __restrict__ unoise,
           _Float16* __restrict__ adj, float* __restrict__ invd0)
{
    const int tid = threadIdx.x;
    const size_t e0 = (size_t)blockIdx.x * GN;       // element base (b*N+i)*N
    const float4* L = (const float4*)logits;         // float4 = 2 elements
    const float4* Uq = (const float4*)unoise;
    const size_t q0 = e0 / 2 + (size_t)tid * 2;
    float4 l01 = L[q0], l23 = L[q0 + 1];
    float4 u01 = Uq[q0], u23 = Uq[q0 + 1];
    float a[8] = {l01.x, l01.y, l01.z, l01.w, l23.x, l23.y, l23.z, l23.w};
    float un[8] = {u01.x, u01.y, u01.z, u01.w, u23.x, u23.y, u23.z, u23.w};
    v4h h4;
    float s = 0.0f;
#pragma unroll
    for (int e = 0; e < 4; ++e) {
        float n0 = -logf(GEPS - logf(un[2 * e] + GEPS));
        float n1 = -logf(GEPS - logf(un[2 * e + 1] + GEPS));
        float z = ((a[2 * e + 1] + n1) - (a[2 * e] + n0)) / GTAU;
        float p = 1.0f / (1.0f + expf(z));           // softmax component 0
        s += p;
        h4[e] = (_Float16)p;
    }
    *(v4h*)(adj + e0 + tid * 4) = h4;

    __shared__ float red[256];
    red[tid] = s;
    __syncthreads();
    for (int st = 128; st > 0; st >>= 1) {
        if (tid < st) red[tid] += red[tid + st];
        __syncthreads();
    }
    if (tid == 0) invd0[blockIdx.x] = 1.0f / (1.0f + red[0]);  // +1: identity
}

// ---------------------------------------------------------------------------
// Kernel 4: column sums -> invd1 (row-degree of adj^T + I)
// ---------------------------------------------------------------------------
__global__ __launch_bounds__(256)
void k_cols(const _Float16* __restrict__ adj, float* __restrict__ invd1)
{
    const int j = blockIdx.x * 256 + threadIdx.x;
    const int b = blockIdx.y;
    float s = 0.0f;
    const _Float16* p = adj + (size_t)b * GN * GN + j;
    for (int i = 0; i < GN; ++i) s += (float)p[(size_t)i * GN];
    invd1[b * GN + j] = 1.0f / (1.0f + s);
}

// ---------------------------------------------------------------------------
// Kernel 5/7: Z = invd .* (X + adj(^T) @ X)   — the diffusion GEMM.
// Block: 256 thr (8 waves), 128x128 output tile, K loop over 1024 in 32-chunks.
// blockIdx.y==1 -> use adj^T (transpose applied when staging to LDS).
// Row-contiguous A staging goes via async global->LDS (ASYNCcnt).
// ---------------------------------------------------------------------------
__global__ __launch_bounds__(256)
void k_diffuse(const _Float16* __restrict__ adj, const _Float16* __restrict__ X,
               const float* __restrict__ invd0, const float* __restrict__ invd1,
               _Float16* __restrict__ Z0, _Float16* __restrict__ Z1)
{
    const int tid = threadIdx.x;
    const int b = blockIdx.z;
    const int trans = blockIdx.y;
    const int m0 = blockIdx.x * 128;
    __shared__ __align__(16) _Float16 At[128 * 40];  // [row][k], stride 40 halves
    __shared__ __align__(16) _Float16 Bt[128 * 40];  // [n][k]  (column-major X)
    const unsigned atBase = lds_off(&At[0]);

    v8f acc[8];
#pragma unroll
    for (int i = 0; i < 8; ++i) acc[i] = (v8f)0.0f;

    const int wave = tid >> 5, lane = tid & 31;
    const int lhalf = (lane < 16) ? 0 : 1;
    const int am = wave * 16 + (lane & 15);
    const int akb = lhalf * 16;         // A: K base 0 / 16 (+8 for hi half)
    const int bn = lane & 15;
    const int bkb = lhalf * 16;         // B: K base 0 / 16, 16 contiguous

#pragma unroll 1
    for (int kt = 0; kt < 32; ++kt) {
        const int k0 = kt * 32;
        if (!trans) {
            // row-major adj tile: pure 16B/lane copies -> async to LDS
#pragma unroll
            for (int it = 0; it < 2; ++it) {
                int idx = it * 256 + tid, row = idx >> 2, kq = idx & 3;
                async_ld_b128(atBase + (unsigned)(row * 40 + kq * 8) * 2,
                              adj + (size_t)(b * GN + m0 + row) * GN + k0 + kq * 8);
            }
        } else {
#pragma unroll
            for (int it = 0; it < 2; ++it) {
                int idx = it * 256 + tid, kk = idx >> 4, c8 = idx & 15;
                uint4 v = *(const uint4*)(adj + (size_t)(b * GN + k0 + kk) * GN + m0 + c8 * 8);
                const _Float16* h = (const _Float16*)&v;
#pragma unroll
                for (int j = 0; j < 8; ++j) At[(c8 * 8 + j) * 40 + kk] = h[j];
            }
        }
#pragma unroll
        for (int it = 0; it < 2; ++it) {            // X tile, stored transposed
            int idx = it * 256 + tid, kk = idx >> 4, c8 = idx & 15;
            uint4 v = *(const uint4*)(X + (size_t)(b * GN + k0 + kk) * TWOU + c8 * 8);
            const _Float16* h = (const _Float16*)&v;
#pragma unroll
            for (int j = 0; j < 8; ++j) Bt[(c8 * 8 + j) * 40 + kk] = h[j];
        }
        if (!trans) wait_asynccnt0();               // our async writes landed
        __syncthreads();
        v16h a;
        {
            v8h lo = *(const v8h*)(&At[am * 40 + akb]);
            v8h hi = *(const v8h*)(&At[am * 40 + akb + 8]);
#pragma unroll
            for (int j = 0; j < 8; ++j) { a[j] = lo[j]; a[8 + j] = hi[j]; }
        }
#pragma unroll
        for (int nf = 0; nf < 8; ++nf) {
            v16h bf = *(const v16h*)(&Bt[(nf * 16 + bn) * 40 + bkb]);
            acc[nf] = __builtin_amdgcn_wmma_f32_16x16x32_f16(
                false, a, false, bf, (short)0, acc[nf], false, false);
        }
        __syncthreads();
    }

    const float* invd = trans ? invd1 : invd0;
    _Float16* Z = trans ? Z1 : Z0;
    const int mb = m0 + wave * 16 + lhalf * 8;      // C/D: lanes>=16 hold M+8
#pragma unroll
    for (int nf = 0; nf < 8; ++nf) {
        const int n = nf * 16 + bn;
#pragma unroll
        for (int r = 0; r < 8; ++r) {
            const int m = mb + r;
            const size_t gi = (size_t)(b * GN + m) * TWOU + n;
            float z = invd[b * GN + m] * ((float)X[gi] + acc[nf][r]);
            Z[gi] = (_Float16)z;
        }
    }
}

// ---------------------------------------------------------------------------
// Kernel 6: value = sigmoid(X1@Wxa + Z0@W0b + Z1@W1b + b); r,u split;
// writes X2[:,64:] = r .* h and u_buf.
// ---------------------------------------------------------------------------
__global__ __launch_bounds__(256)
void k_value(const _Float16* __restrict__ X1, const _Float16* __restrict__ Z0,
             const _Float16* __restrict__ Z1,
             const _Float16* __restrict__ Wxa, const _Float16* __restrict__ W0b,
             const _Float16* __restrict__ W1b, const float* __restrict__ bsum,
             const float* __restrict__ hx,
             _Float16* __restrict__ X2, float* __restrict__ u_buf)
{
    const int tid = threadIdx.x;
    const int b = blockIdx.y;
    const int m0 = blockIdx.x * 128;
    __shared__ __align__(16) _Float16 At[128 * 40];
    __shared__ __align__(16) _Float16 Bt[128 * 40];
    const unsigned atBase = lds_off(&At[0]);

    v8f acc[8];
#pragma unroll
    for (int i = 0; i < 8; ++i) acc[i] = (v8f)0.0f;

    const int wave = tid >> 5, lane = tid & 31;
    const int lhalf = (lane < 16) ? 0 : 1;
    const int am = wave * 16 + (lane & 15);
    const int akb = lhalf * 16;
    const int bn = lane & 15;
    const int bkb = lhalf * 16;

    const _Float16* Ains[3] = {X1, Z0, Z1};
    const _Float16* Bws[3]  = {Wxa, W0b, W1b};

#pragma unroll 1
    for (int seg = 0; seg < 3; ++seg) {
        const _Float16* Ain = Ains[seg];
        const _Float16* Bw = Bws[seg];
#pragma unroll 1
        for (int kc = 0; kc < 4; ++kc) {
            const int k0 = kc * 32;
#pragma unroll
            for (int it = 0; it < 2; ++it) {        // A tile: async row copies
                int idx = it * 256 + tid, row = idx >> 2, kq = idx & 3;
                async_ld_b128(atBase + (unsigned)(row * 40 + kq * 8) * 2,
                              Ain + (size_t)(b * GN + m0 + row) * TWOU + k0 + kq * 8);
            }
#pragma unroll
            for (int it = 0; it < 2; ++it) {        // weight tile -> column-major
                int idx = it * 256 + tid, kk = idx >> 4, c8 = idx & 15;
                uint4 v = *(const uint4*)(Bw + (size_t)(k0 + kk) * TWOU + c8 * 8);
                const _Float16* h = (const _Float16*)&v;
#pragma unroll
                for (int j = 0; j < 8; ++j) Bt[(c8 * 8 + j) * 40 + kk] = h[j];
            }
            wait_asynccnt0();
            __syncthreads();
            v16h a;
            {
                v8h lo = *(const v8h*)(&At[am * 40 + akb]);
                v8h hi = *(const v8h*)(&At[am * 40 + akb + 8]);
#pragma unroll
                for (int j = 0; j < 8; ++j) { a[j] = lo[j]; a[8 + j] = hi[j]; }
            }
#pragma unroll
            for (int nf = 0; nf < 8; ++nf) {
                v16h bf = *(const v16h*)(&Bt[(nf * 16 + bn) * 40 + bkb]);
                acc[nf] = __builtin_amdgcn_wmma_f32_16x16x32_f16(
                    false, a, false, bf, (short)0, acc[nf], false, false);
            }
            __syncthreads();
        }
    }

    const int mb = m0 + wave * 16 + lhalf * 8;
#pragma unroll
    for (int nf = 0; nf < 8; ++nf) {
        const int n = nf * 16 + bn;
#pragma unroll
        for (int r = 0; r < 8; ++r) {
            const int m = mb + r;
            float v = 1.0f / (1.0f + expf(-(acc[nf][r] + bsum[n])));
            if (n < GU) {   // r-gate -> X2 second half = r * h
                float hv = hx[(size_t)b * (GN * GU) + m * GU + n];
                X2[(size_t)(b * GN + m) * TWOU + GU + n] = (_Float16)(v * hv);
            } else {        // u-gate
                u_buf[(size_t)(b * GN + m) * GU + (n - GU)] = v;
            }
        }
    }
}

// ---------------------------------------------------------------------------
// Kernel 8: c = tanh(X2@Wcxa + Z0'@Wc0b + Z1'@Wc1b + bc);
// out = u*h + (1-u)*c.  Output tile 128 x 64.
// ---------------------------------------------------------------------------
__global__ __launch_bounds__(256)
void k_out(const _Float16* __restrict__ X2, const _Float16* __restrict__ Z0,
           const _Float16* __restrict__ Z1,
           const _Float16* __restrict__ Wcxa, const _Float16* __restrict__ Wc0b,
           const _Float16* __restrict__ Wc1b, const float* __restrict__ bcsum,
           const float* __restrict__ hx, const float* __restrict__ u_buf,
           float* __restrict__ out)
{
    const int tid = threadIdx.x;
    const int b = blockIdx.y;
    const int m0 = blockIdx.x * 128;
    __shared__ __align__(16) _Float16 At[128 * 40];
    __shared__ __align__(16) _Float16 Bt[64 * 40];
    const unsigned atBase = lds_off(&At[0]);

    v8f acc[4];
#pragma unroll
    for (int i = 0; i < 4; ++i) acc[i] = (v8f)0.0f;

    const int wave = tid >> 5, lane = tid & 31;
    const int lhalf = (lane < 16) ? 0 : 1;
    const int am = wave * 16 + (lane & 15);
    const int akb = lhalf * 16;
    const int bn = lane & 15;
    const int bkb = lhalf * 16;

    const _Float16* Ains[3] = {X2, Z0, Z1};
    const _Float16* Bws[3]  = {Wcxa, Wc0b, Wc1b};

#pragma unroll 1
    for (int seg = 0; seg < 3; ++seg) {
        const _Float16* Ain = Ains[seg];
        const _Float16* Bw = Bws[seg];
#pragma unroll 1
        for (int kc = 0; kc < 4; ++kc) {
            const int k0 = kc * 32;
#pragma unroll
            for (int it = 0; it < 2; ++it) {        // A tile: async row copies
                int idx = it * 256 + tid, row = idx >> 2, kq = idx & 3;
                async_ld_b128(atBase + (unsigned)(row * 40 + kq * 8) * 2,
                              Ain + (size_t)(b * GN + m0 + row) * TWOU + k0 + kq * 8);
            }
            {   // weight tile 32 x 64 -> Bt[n][k]
                int kk = tid >> 3, c8 = tid & 7;
                uint4 v = *(const uint4*)(Bw + (size_t)(k0 + kk) * GU + c8 * 8);
                const _Float16* h = (const _Float16*)&v;
#pragma unroll
                for (int j = 0; j < 8; ++j) Bt[(c8 * 8 + j) * 40 + kk] = h[j];
            }
            wait_asynccnt0();
            __syncthreads();
            v16h a;
            {
                v8h lo = *(const v8h*)(&At[am * 40 + akb]);
                v8h hi = *(const v8h*)(&At[am * 40 + akb + 8]);
#pragma unroll
                for (int j = 0; j < 8; ++j) { a[j] = lo[j]; a[8 + j] = hi[j]; }
            }
#pragma unroll
            for (int nf = 0; nf < 4; ++nf) {
                v16h bf = *(const v16h*)(&Bt[(nf * 16 + bn) * 40 + bkb]);
                acc[nf] = __builtin_amdgcn_wmma_f32_16x16x32_f16(
                    false, a, false, bf, (short)0, acc[nf], false, false);
            }
            __syncthreads();
        }
    }

    const int mb = m0 + wave * 16 + lhalf * 8;
#pragma unroll
    for (int nf = 0; nf < 4; ++nf) {
        const int n = nf * 16 + bn;
#pragma unroll
        for (int r = 0; r < 8; ++r) {
            const int m = mb + r;
            float cc = tanhf(acc[nf][r] + bcsum[n]);
            size_t gi = (size_t)b * (GN * GU) + m * GU + n;
            float uu = u_buf[(size_t)(b * GN + m) * GU + n];
            out[gi] = uu * hx[gi] + (1.0f - uu) * cc;
        }
    }
}

// ---------------------------------------------------------------------------
extern "C" void kernel_launch(void* const* d_in, const int* in_sizes, int n_in,
                              void* d_out, int out_size, void* d_ws, size_t ws_size,
                              hipStream_t stream)
{
    (void)in_sizes; (void)n_in; (void)out_size; (void)ws_size;
    const float* logits = (const float*)d_in[0];
    const float* unoise = (const float*)d_in[1];
    const float* inputs = (const float*)d_in[2];
    const float* hx     = (const float*)d_in[3];
    const float* W0  = (const float*)d_in[4];
    const float* b0  = (const float*)d_in[5];
    const float* W1  = (const float*)d_in[6];
    const float* b1  = (const float*)d_in[7];
    const float* Wc0 = (const float*)d_in[8];
    const float* bc0 = (const float*)d_in[9];
    const float* Wc1 = (const float*)d_in[10];
    const float* bc1 = (const float*)d_in[11];
    float* out = (float*)d_out;

    char* w = (char*)d_ws;
    size_t off = 0;
    auto alloc = [&](size_t bytes) -> void* {
        void* p = w + off;
        off += (bytes + 255) & ~(size_t)255;
        return p;
    };
    _Float16* adj  = (_Float16*)alloc((size_t)GB * GN * GN * 2);   // 32 MB
    _Float16* X1   = (_Float16*)alloc((size_t)GB * GN * TWOU * 2); // 4 MB
    _Float16* X2   = (_Float16*)alloc((size_t)GB * GN * TWOU * 2);
    _Float16* Z0   = (_Float16*)alloc((size_t)GB * GN * TWOU * 2);
    _Float16* Z1   = (_Float16*)alloc((size_t)GB * GN * TWOU * 2);
    float* u_buf   = (float*)alloc((size_t)GB * GN * GU * 4);      // 4 MB
    float* invd0   = (float*)alloc((size_t)GB * GN * 4);
    float* invd1   = (float*)alloc((size_t)GB * GN * 4);
    _Float16* Wxa  = (_Float16*)alloc(128 * 128 * 2);
    _Float16* W0b  = (_Float16*)alloc(128 * 128 * 2);
    _Float16* W1b  = (_Float16*)alloc(128 * 128 * 2);
    _Float16* Wcxa = (_Float16*)alloc(128 * 64 * 2);
    _Float16* Wc0b = (_Float16*)alloc(128 * 64 * 2);
    _Float16* Wc1b = (_Float16*)alloc(128 * 64 * 2);
    float* bsum    = (float*)alloc(128 * 4);
    float* bcsum   = (float*)alloc(64 * 4);

    k_prep<<<1, 256, 0, stream>>>(W0, b0, W1, b1, Wc0, bc0, Wc1, bc1,
                                  Wxa, W0b, W1b, Wcxa, Wc0b, Wc1b, bsum, bcsum);
    k_pack<<<(GB * GN * GU) / 256, 256, 0, stream>>>(inputs, hx, X1, X2);
    k_adj<<<GB * GN, 256, 0, stream>>>(logits, unoise, adj, invd0);
    k_cols<<<dim3(GN / 256, GB), 256, 0, stream>>>(adj, invd1);
    // gate-pass diffusion: Z0 = A0@X1, Z1 = A1@X1
    k_diffuse<<<dim3(GN / 128, 2, GB), 256, 0, stream>>>(adj, X1, invd0, invd1, Z0, Z1);
    k_value<<<dim3(GN / 128, GB), 256, 0, stream>>>(X1, Z0, Z1, Wxa, W0b, W1b,
                                                    bsum, hx, X2, u_buf);
    // candidate-pass diffusion (reuses Z0/Z1): Z0 = A0@X2, Z1 = A1@X2
    k_diffuse<<<dim3(GN / 128, 2, GB), 256, 0, stream>>>(adj, X2, invd0, invd1, Z0, Z1);
    k_out<<<dim3(GN / 128, GB), 256, 0, stream>>>(X2, Z0, Z1, Wcxa, Wc0b, Wc1b,
                                                  bcsum, hx, u_buf, out);
}
